// MultiHeadAttention_66254165508269
// MI455X (gfx1250) — matrix operand order repaired and checked by reading, not compiled, not run on hardware
//
#include <hip/hip_runtime.h>

typedef __attribute__((ext_vector_type(16))) _Float16 v16h;
typedef __attribute__((ext_vector_type(8)))  float    v8f;

#define D_MODEL 1024
#define SEQ     2048
#define BATCH   2
#define NH      16
#define HD      64
#define MTOT    (BATCH * SEQ)   // 4096 rows

// ---------------- CDNA5 async memory -> LDS path ----------------
#if __has_builtin(__builtin_amdgcn_global_load_async_to_lds_b128)
#define ASYNC_BUILTIN 1
#else
#define ASYNC_BUILTIN 0
#endif

typedef int v4i_vs __attribute__((vector_size(16)));
typedef __attribute__((address_space(1))) v4i_vs as1_v4i;
typedef __attribute__((address_space(3))) v4i_vs as3_v4i;

__device__ __forceinline__ void async_ld_b128(const void* g, void* l) {
#if ASYNC_BUILTIN
  __builtin_amdgcn_global_load_async_to_lds_b128((as1_v4i*)g, (as3_v4i*)l, 0, 0);
#else
  // vdst = LDS byte address, vaddr = 64-bit global address (GV mode)
  asm volatile("global_load_async_to_lds_b128 %0, %1, off"
               :: "v"((unsigned)(unsigned long long)l), "v"(g)
               : "memory");
#endif
}

__device__ __forceinline__ void wait_async0() {
#if __has_builtin(__builtin_amdgcn_s_wait_asynccnt)
  __builtin_amdgcn_s_wait_asynccnt(0);
#else
  asm volatile("s_wait_asynccnt 0x0" ::: "memory");
#endif
}
__device__ __forceinline__ void wait_async4() {
#if __has_builtin(__builtin_amdgcn_s_wait_asynccnt)
  __builtin_amdgcn_s_wait_asynccnt(4);
#else
  asm volatile("s_wait_asynccnt 0x4" ::: "memory");
#endif
}

// ---------- fragment loaders (wave32 WMMA 16x16x32 f16 layouts) ----------
// A-matrix 16x32: lane l (row = l&15): K chunks [seg..seg+7], [seg+16..seg+23], seg=(l&16)?8:0
__device__ __forceinline__ v16h load_fragA(const _Float16* p /* row base + seg */) {
  union { uint4 u[2]; v16h h; } u;
  u.u[0] = *(const uint4*)(p);
  u.u[1] = *(const uint4*)(p + 16);
  return u.h;
}
// B-matrix 32x16: lane l (col = l&15): K chunks [seg..seg+7], [seg+16..seg+23], seg=(l&16)?8:0
__device__ __forceinline__ v16h load_fragB(const _Float16* rowbase, int lane) {
  const _Float16* p = rowbase + ((lane & 16) ? 8 : 0);
  union { uint4 u[2]; v16h h; } u;
  u.u[0] = *(const uint4*)(p);
  u.u[1] = *(const uint4*)(p + 16);
  return u.h;
}

// ---------- conversion kernels ----------
__global__ void __launch_bounds__(256) k_cvt_f16(const float* __restrict__ in,
                                                 _Float16* __restrict__ out, int n) {
  int i = blockIdx.x * 256 + threadIdx.x;
  if (i < n) out[i] = (_Float16)in[i];
}

// out[n*1024 + k] = in[k*1024 + n]  (W^T, f16)
__global__ void __launch_bounds__(256) k_transpose_w(const float* __restrict__ in,
                                                     _Float16* __restrict__ out) {
  int i = blockIdx.x * 256 + threadIdx.x;
  int n = i >> 10;
  int k = i & 1023;
  out[i] = (_Float16)in[k * D_MODEL + n];
}

// ---------- WMMA GEMM:  C[M=4096, N=1024] = A[M,K=1024] * BT[N,K]^T + bias ----------
// mode 0: Q -> f16 [bh][s][hd], scaled by 1/sqrt(HD)
// mode 1: K -> f16 [bh][s][hd]
// mode 2: V -> f16 [bh][hd][s]
// mode 3: O -> f32 [t][n]
__global__ void __launch_bounds__(128)
k_gemm_wmma(const _Float16* __restrict__ A, const _Float16* __restrict__ BT,
            const float* __restrict__ bias, void* __restrict__ outp, int mode) {
  __shared__ _Float16 Bs[2][64][72];   // double-buffered: 64 N-rows x 64 K (+8 pad)

  const int lane = threadIdx.x & 31;
  const int wv   = threadIdx.x >> 5;
  const int mbase = blockIdx.y * 64 + wv * 16;
  const int nbase = blockIdx.x * 64;
  const int seg = (lane & 16) ? 8 : 0;
  const int rl  = lane & 15;

  // staging assignment: thread -> (row, 32-half chunk); 4 async b128 per thread per tile
  const int sr = threadIdx.x >> 1;
  const int sc = (threadIdx.x & 1) * 32;

  v8f acc[4] = {};
  const _Float16* arow = A + (size_t)(mbase + rl) * D_MODEL + seg;

  // prologue: stage K-slab 0 into buffer 0
  {
    const _Float16* src = BT + (size_t)(nbase + sr) * D_MODEL + sc;
#pragma unroll
    for (int j = 0; j < 4; ++j) async_ld_b128(src + j * 8, &Bs[0][sr][sc + j * 8]);
  }

  const int nIter = D_MODEL / 64;   // 16
  for (int it = 0; it < nIter; ++it) {
    const int kb  = it * 64;
    const int buf = it & 1;
    if (it + 1 < nIter) {   // prefetch next slab into the other buffer
      const _Float16* src = BT + (size_t)(nbase + sr) * D_MODEL + (kb + 64) + sc;
#pragma unroll
      for (int j = 0; j < 4; ++j) async_ld_b128(src + j * 8, &Bs[buf ^ 1][sr][sc + j * 8]);
      wait_async4();        // previous slab's 4 loads complete; new 4 in flight
    } else {
      wait_async0();
    }
    __syncthreads();        // slab visible to all waves

    v16h af0 = load_fragA(arow + kb);
    v16h af1 = load_fragA(arow + kb + 32);
#pragma unroll
    for (int nt = 0; nt < 4; ++nt) {
      v16h bf0 = load_fragB(&Bs[buf][nt * 16 + rl][0], lane);
      acc[nt] = __builtin_amdgcn_wmma_f32_16x16x32_f16(false, af0, false, bf0,
                                                       (short)0, acc[nt], false, false);
      v16h bf1 = load_fragB(&Bs[buf][nt * 16 + rl][32], lane);
      acc[nt] = __builtin_amdgcn_wmma_f32_16x16x32_f16(false, af1, false, bf1,
                                                       (short)0, acc[nt], false, false);
    }
    __syncthreads();        // all reads done before buffer is overwritten
  }

  const int hi8 = (lane & 16) ? 8 : 0;
#pragma unroll
  for (int nt = 0; nt < 4; ++nt) {
#pragma unroll
    for (int r = 0; r < 8; ++r) {
      int m = mbase + hi8 + r;
      int n = nbase + nt * 16 + rl;
      float v = acc[nt][r] + bias[n];
      if (mode == 3) {
        ((float*)outp)[(size_t)m * D_MODEL + n] = v;
      } else {
        int b = m >> 11, s = m & 2047;
        int h = n >> 6,  d = n & 63;
        int bh = b * NH + h;
        _Float16* o = (_Float16*)outp;
        if (mode == 0)      o[((size_t)bh * SEQ + s) * HD + d] = (_Float16)(v * 0.125f);
        else if (mode == 1) o[((size_t)bh * SEQ + s) * HD + d] = (_Float16)v;
        else                o[((size_t)bh * HD + d) * SEQ + s] = (_Float16)v;
      }
    }
  }
}

// ---------- Flash attention: block = (b,h) x 64 query rows; 4 waves x 16 rows ----------
__global__ void __launch_bounds__(128)
k_flash_attn(const _Float16* __restrict__ Q, const _Float16* __restrict__ Kb,
             const _Float16* __restrict__ Vt, _Float16* __restrict__ C) {
  __shared__ _Float16 Ks[2][32][72];     // kv x hd (+8 pad), double buffered
  __shared__ _Float16 Vs[2][64][40];     // hd x kv (+8 pad), double buffered
  __shared__ _Float16 Ps[4][16][40];     // per-wave P tile 16 x 32 (+8 pad)

  const int lane = threadIdx.x & 31;
  const int wv   = threadIdx.x >> 5;
  const int bh   = blockIdx.y;
  const int qbase = blockIdx.x * 64 + wv * 16;
  const int seg = (lane & 16) ? 8 : 0;
  const int rl  = lane & 15;
  const int hi8 = (lane & 16) ? 8 : 0;

  const _Float16* qrow = Q + ((size_t)bh * SEQ + qbase + rl) * HD + seg;
  v16h qf0 = load_fragA(qrow);        // hd 0..31
  v16h qf1 = load_fragA(qrow + 32);   // hd 32..63

  v8f o[4] = {};
  float mrow[8], lrow[8];
#pragma unroll
  for (int r = 0; r < 8; ++r) { mrow[r] = -1e30f; lrow[r] = 0.f; }

  const _Float16* Kbase = Kb + (size_t)bh * SEQ * HD;
  const _Float16* Vbase = Vt + (size_t)bh * HD * SEQ;

  // staging assignment (4 async b128 per thread per kv-tile)
  const int kr = threadIdx.x >> 2, kc = (threadIdx.x & 3) * 16;
  const int vr = threadIdx.x >> 1, vc = (threadIdx.x & 1) * 16;

  // prologue: stage tile 0
  {
    const _Float16* ksrc = Kbase + (size_t)kr * HD + kc;
    async_ld_b128(ksrc,     &Ks[0][kr][kc]);
    async_ld_b128(ksrc + 8, &Ks[0][kr][kc + 8]);
    const _Float16* vsrc = Vbase + (size_t)vr * SEQ + vc;
    async_ld_b128(vsrc,     &Vs[0][vr][vc]);
    async_ld_b128(vsrc + 8, &Vs[0][vr][vc + 8]);
  }

  const int nIter = SEQ / 32;   // 64
  for (int it = 0; it < nIter; ++it) {
    const int buf = it & 1;
    if (it + 1 < nIter) {
      const int kvn = (it + 1) * 32;
      const _Float16* ksrc = Kbase + (size_t)(kvn + kr) * HD + kc;
      async_ld_b128(ksrc,     &Ks[buf ^ 1][kr][kc]);
      async_ld_b128(ksrc + 8, &Ks[buf ^ 1][kr][kc + 8]);
      const _Float16* vsrc = Vbase + (size_t)vr * SEQ + kvn + vc;
      async_ld_b128(vsrc,     &Vs[buf ^ 1][vr][vc]);
      async_ld_b128(vsrc + 8, &Vs[buf ^ 1][vr][vc + 8]);
      wait_async4();
    } else {
      wait_async0();
    }
    __syncthreads();

    // scores S[16 q x 32 kv] = Q(16x64) * K^T(64x32)  (Q pre-scaled by 1/8)
    v8f s[2];
#pragma unroll
    for (int nt = 0; nt < 2; ++nt) {
      v8f z = {};
      v16h b0 = load_fragB(&Ks[buf][nt * 16 + rl][0], lane);
      s[nt] = __builtin_amdgcn_wmma_f32_16x16x32_f16(false, qf0, false, b0, (short)0, z, false, false);
      v16h b1 = load_fragB(&Ks[buf][nt * 16 + rl][32], lane);
      s[nt] = __builtin_amdgcn_wmma_f32_16x16x32_f16(false, qf1, false, b1, (short)0, s[nt], false, false);
    }

    // online softmax (rows live in VGPR index; 16 lanes of each half share a row)
#pragma unroll
    for (int r = 0; r < 8; ++r) {
      float mx = fmaxf(s[0][r], s[1][r]);
      mx = fmaxf(mx, __shfl_xor(mx, 1, 32));
      mx = fmaxf(mx, __shfl_xor(mx, 2, 32));
      mx = fmaxf(mx, __shfl_xor(mx, 4, 32));
      mx = fmaxf(mx, __shfl_xor(mx, 8, 32));
      float mnew = fmaxf(mrow[r], mx);
      float corr = __expf(mrow[r] - mnew);
      mrow[r] = mnew;
      float p0 = __expf(s[0][r] - mnew);
      float p1 = __expf(s[1][r] - mnew);
      s[0][r] = p0; s[1][r] = p1;
      float rs = p0 + p1;
      rs += __shfl_xor(rs, 1, 32);
      rs += __shfl_xor(rs, 2, 32);
      rs += __shfl_xor(rs, 4, 32);
      rs += __shfl_xor(rs, 8, 32);
      lrow[r] = lrow[r] * corr + rs;
#pragma unroll
      for (int dt = 0; dt < 4; ++dt) o[dt][r] = o[dt][r] * corr;
    }

    // C-layout P -> LDS -> A-layout fragment (per-wave; LDS in-order per wave)
#pragma unroll
    for (int nt = 0; nt < 2; ++nt)
#pragma unroll
      for (int r = 0; r < 8; ++r)
        Ps[wv][hi8 + r][nt * 16 + rl] = (_Float16)s[nt][r];
    asm volatile("" ::: "memory");

    v16h pf = load_fragA(&Ps[wv][rl][seg]);
#pragma unroll
    for (int dt = 0; dt < 4; ++dt) {
      v16h bf = load_fragB(&Vs[buf][dt * 16 + rl][0], lane);
      o[dt] = __builtin_amdgcn_wmma_f32_16x16x32_f16(false, pf, false, bf, (short)0, o[dt], false, false);
    }
    __syncthreads();   // all reads done before the buffer is overwritten
  }

  // normalize and store ctx (f16, [B,S,D] with col = h*64 + d)
  const int b = bh >> 4, h = bh & 15;
#pragma unroll
  for (int r = 0; r < 8; ++r) {
    float inv = 1.0f / lrow[r];
    int s_idx = qbase + hi8 + r;
    size_t rowoff = ((size_t)(b * SEQ + s_idx)) * D_MODEL + h * HD;
#pragma unroll
    for (int dt = 0; dt < 4; ++dt)
      C[rowoff + dt * 16 + rl] = (_Float16)(o[dt][r] * inv);
  }
}

// ---------- launcher ----------
extern "C" void kernel_launch(void* const* d_in, const int* in_sizes, int n_in,
                              void* d_out, int out_size, void* d_ws, size_t ws_size,
                              hipStream_t stream) {
  (void)in_sizes; (void)n_in; (void)out_size; (void)ws_size;
  const float* x  = (const float*)d_in[0];
  const float* Wq = (const float*)d_in[1];
  const float* bq = (const float*)d_in[2];
  const float* Wk = (const float*)d_in[3];
  const float* bk = (const float*)d_in[4];
  const float* Wv = (const float*)d_in[5];
  const float* bv = (const float*)d_in[6];
  const float* Wo = (const float*)d_in[7];
  const float* bo = (const float*)d_in[8];

  const size_t NX = (size_t)MTOT * D_MODEL;     // 4M elements
  const size_t NW = (size_t)D_MODEL * D_MODEL;  // 1M elements

  _Float16* ws  = (_Float16*)d_ws;
  _Float16* x16 = ws;            // 4M
  _Float16* WqT = x16 + NX;      // 1M
  _Float16* WkT = WqT + NW;
  _Float16* WvT = WkT + NW;
  _Float16* WoT = WvT + NW;
  _Float16* Qb  = WoT + NW;      // 4M  [bh][s][hd], pre-scaled
  _Float16* Kb  = Qb + NX;       // 4M  [bh][s][hd]
  _Float16* Vb  = Kb + NX;       // 4M  [bh][hd][s]
  _Float16* Cb  = Vb + NX;       // 4M  ctx f16 [t][n]

  k_cvt_f16<<<(int)(NX / 256), 256, 0, stream>>>(x, x16, (int)NX);
  k_transpose_w<<<(int)(NW / 256), 256, 0, stream>>>(Wq, WqT);
  k_transpose_w<<<(int)(NW / 256), 256, 0, stream>>>(Wk, WkT);
  k_transpose_w<<<(int)(NW / 256), 256, 0, stream>>>(Wv, WvT);
  k_transpose_w<<<(int)(NW / 256), 256, 0, stream>>>(Wo, WoT);

  dim3 gg(D_MODEL / 64, MTOT / 64);   // (16, 64) blocks, 128 threads
  k_gemm_wmma<<<gg, 128, 0, stream>>>(x16, WqT, bq, Qb, 0);
  k_gemm_wmma<<<gg, 128, 0, stream>>>(x16, WkT, bk, Kb, 1);
  k_gemm_wmma<<<gg, 128, 0, stream>>>(x16, WvT, bv, Vb, 2);

  k_flash_attn<<<dim3(SEQ / 64, BATCH * NH), 128, 0, stream>>>(Qb, Kb, Vb, Cb);

  k_gemm_wmma<<<gg, 128, 0, stream>>>(Cb, WoT, bo, d_out, 3);
}
